// Block_46600395161878
// MI455X (gfx1250) — matrix operand order repaired and checked by reading, not compiled, not run on hardware
//
#include <hip/hip_runtime.h>
#include <hip/hip_bf16.h>
#include <math.h>

// ---------------------------------------------------------------------------
// Transformer block forward for MI455X (gfx1250), bf16 WMMA + fp32 accum.
//   B=4, T=2048, C=1024, H=16, hs=64.  ~274 GFLOP, compute-bound =>
//   all GEMM-like work on v_wmma_f32_16x16x32_bf16; global->LDS staging on
//   the CDNA5 async path (global_load_async_to_lds_b128 / ASYNCcnt) and the
//   Tensor Data Mover (tensor_load_to_lds / TENSORcnt), double-buffered.
// ---------------------------------------------------------------------------

#define D_MODEL 1024
#define N_HEAD  16
#define HEAD_SZ 64
#define SEQ     2048
#define BATCH   4
#define MROWS   (BATCH * SEQ)        // 8192
#define C3      (3 * D_MODEL)        // 3072
#define FF      (4 * D_MODEL)        // 4096

typedef __attribute__((ext_vector_type(16))) __bf16 v16bf;
typedef __attribute__((ext_vector_type(8)))  float  v8f;
typedef __attribute__((ext_vector_type(4)))  unsigned v4u;
typedef __attribute__((ext_vector_type(8)))  int      v8i;
typedef __attribute__((ext_vector_type(4)))  int      v4i;

union FragU { v16bf bf; unsigned int u[8]; };

#if __has_builtin(__builtin_amdgcn_tensor_load_to_lds)
#define HAVE_TDM 1
#else
#define HAVE_TDM 0
#endif

// ---- CDNA5 async global->LDS copy (16B per lane), ASYNCcnt-tracked --------
__device__ inline void async_copy16(const __bf16* g, const __bf16* l) {
  // generic shared pointer: low 32 bits are the LDS byte offset (ISA 10.2)
  unsigned lds = (unsigned)(size_t)l;
  asm volatile("global_load_async_to_lds_b128 %0, %1, off"
               :: "v"(lds), "v"(g) : "memory");
}
__device__ inline void wait_async0() {
  asm volatile("s_wait_asynccnt 0x0" ::: "memory");
}
__device__ inline void wait_tensor0() {
#if __has_builtin(__builtin_amdgcn_s_wait_tensorcnt)
  __builtin_amdgcn_s_wait_tensorcnt(0);
#else
  asm volatile("s_wait_tensorcnt 0x0" ::: "memory");
#endif
}

// ---- Tensor Data Mover: 2D tile (rows x cols of bf16) global -> LDS, with
// LDS padding: pad_interval=16 dwords (one 32-elem row), pad_amount=4 dwords
// (8 elems) -> reproduces the LDA/LDBT = 40-element padded rows in hardware.
__device__ inline void tdm_load_2d_pad(unsigned lds_byte, const void* gptr,
                                       unsigned tile_cols, unsigned tile_rows,
                                       unsigned row_stride_elems) {
#if HAVE_TDM
  unsigned long long ga = (unsigned long long)gptr;
  v4u g0;
  g0[0] = 1u;                                   // count=1, user descriptor
  g0[1] = lds_byte;                             // lds_addr
  g0[2] = (unsigned)ga;                         // global_addr[31:0]
  g0[3] = (unsigned)((ga >> 32) & 0x1ffffffu)   // global_addr[56:32]
          | 0x80000000u;                        // type=2 ("image")
  v8i g1;
  g1[0] = (int)((1u << 16)        // data_size = 2 bytes
                | (1u << 20)      // pad_enable
                | (3u << 22)      // pad_interval code 3 -> 16 dwords
                | (3u << 25));    // pad_amount  code 3 -> 4 dwords
  g1[1] = (int)((tile_cols & 0xffffu) << 16);           // tensor_dim0 lo
  g1[2] = (int)((tile_cols >> 16) |
                ((tile_rows & 0xffffu) << 16));         // dim0 hi | dim1 lo
  g1[3] = (int)((tile_rows >> 16) | (tile_cols << 16)); // dim1 hi | tile_dim0
  g1[4] = (int)(tile_rows & 0xffffu);                   // tile_dim1
  g1[5] = (int)row_stride_elems;                        // tensor_dim0_stride
  g1[6] = 0; g1[7] = 0;
  v4i z = {0, 0, 0, 0};
#if defined(__clang_major__) && (__clang_major__ >= 23)
  v8i z8 = {0, 0, 0, 0, 0, 0, 0, 0};
  __builtin_amdgcn_tensor_load_to_lds(g0, g1, z, z, z8, 0);
#else
  __builtin_amdgcn_tensor_load_to_lds(g0, g1, z, z, 0);
#endif
#endif
}

// ---- WMMA fragment loaders (CDNA5 wave32 layouts, CDNA5 ISA) ----
__device__ inline v16bf load_a_frag(const __bf16* base, int ldm) {
  int lane = threadIdx.x & 31;
  int r = lane & 15, g = lane >> 4;
  FragU a;
#pragma unroll
  for (int v = 0; v < 8; ++v) {
    int k = ((v >> 2) << 4) + (g << 3) + ((v & 3) << 1);
    a.u[v] = *(const unsigned int*)(base + r * ldm + k);
  }
  return a.bf;
}
__device__ inline v16bf load_b_frag(const __bf16* baseT, int ldm) {
  int lane = threadIdx.x & 31;
  int n = lane & 15, g = lane >> 4;
  FragU b;
#pragma unroll
  for (int v = 0; v < 8; ++v) {
    int k = ((v >> 2) << 4) + (g << 3) + ((v & 3) << 1);
    b.u[v] = *(const unsigned int*)(baseT + n * ldm + k);
  }
  return b.bf;
}

__device__ inline v8f wmma_bf16(v16bf a, v16bf b, v8f c) {
  return __builtin_amdgcn_wmma_f32_16x16x32_bf16(
      false, a, false, b, (short)0, c, false, false);
}

__device__ inline float gelu_f(float x) {
  return 0.5f * x * (1.0f + erff(x * 0.70710678118654752f));
}
__device__ inline float rmax16(float v) {
#pragma unroll
  for (int m = 1; m < 16; m <<= 1) v = fmaxf(v, __shfl_xor(v, m, 32));
  return v;
}
__device__ inline float rsum16(float v) {
#pragma unroll
  for (int m = 1; m < 16; m <<= 1) v += __shfl_xor(v, m, 32);
  return v;
}

// ---------------------------------------------------------------------------
// Weight repack fp32 -> bf16
// ---------------------------------------------------------------------------
__global__ void cvt_bf16_kernel(const float* __restrict__ in,
                                __bf16* __restrict__ out, int n) {
  int i = blockIdx.x * 256 + threadIdx.x;
  if (i < n) out[i] = (__bf16)in[i];
}

__global__ void pack_qkv_kernel(const float* __restrict__ Wq,
                                const float* __restrict__ Wk,
                                const float* __restrict__ Wv,
                                __bf16* __restrict__ out) {
  int i = blockIdx.x * 256 + threadIdx.x;          // 1024*3072 total
  int col = i % C3, c = i / C3;
  int which = col >> 10, hc = col & 1023;
  int h = hc >> 6, s = hc & 63;
  const float* W = (which == 0) ? Wq : (which == 1) ? Wk : Wv;
  out[i] = (__bf16)W[((size_t)h * D_MODEL + c) * HEAD_SZ + s];
}

// ---------------------------------------------------------------------------
// LayerNorm: one block per row (1024 cols, 256 threads = 8 waves)
// ---------------------------------------------------------------------------
__global__ void __launch_bounds__(256) ln_kernel(const float* __restrict__ x,
                                                 const float* __restrict__ g,
                                                 const float* __restrict__ be,
                                                 __bf16* __restrict__ out) {
  int row = blockIdx.x;
  const float* xr = x + (size_t)row * D_MODEL;
  float s = 0.f, s2 = 0.f;
  for (int i = threadIdx.x; i < D_MODEL; i += 256) {
    float v = xr[i]; s += v; s2 += v * v;
  }
#pragma unroll
  for (int off = 16; off >= 1; off >>= 1) {
    s += __shfl_down(s, off, 32);
    s2 += __shfl_down(s2, off, 32);
  }
  __shared__ float red[2][8];
  int wid = threadIdx.x >> 5, lane = threadIdx.x & 31;
  if (lane == 0) { red[0][wid] = s; red[1][wid] = s2; }
  __syncthreads();
  if (threadIdx.x == 0) {
    float a = 0.f, b = 0.f;
#pragma unroll
    for (int i = 0; i < 8; ++i) { a += red[0][i]; b += red[1][i]; }
    red[0][0] = a; red[1][0] = b;
  }
  __syncthreads();
  float mean = red[0][0] * (1.0f / D_MODEL);
  float var = red[1][0] * (1.0f / D_MODEL) - mean * mean;
  float rstd = rsqrtf(var + 1e-5f);
  __bf16* orow = out + (size_t)row * D_MODEL;
  for (int i = threadIdx.x; i < D_MODEL; i += 256)
    orow[i] = (__bf16)((xr[i] - mean) * rstd * g[i] + be[i]);
}

// ---------------------------------------------------------------------------
// Tiled bf16 GEMM: C = A[MxK]*B[KxN] (+bias)(+gelu)(+res), out bf16 or f32.
// 128x64 block tile, 8 waves, each wave 32x32 (2x2 register blocking) ->
// 4 WMMAs per BK=32 step.  Double-buffered LDS: the NEXT A tile is staged by
// the Tensor Data Mover (one wave issues the descriptor, TENSORcnt) while the
// current one is consumed; B tile is loaded early to VGPRs and
// transpose-scattered after the WMMAs.
// ---------------------------------------------------------------------------
#define BM 128
#define BN 64
#define BK 32
#define LDA 40   // BK + 8 pad == TDM pad_interval/amount above
#define LDBT 40

template <bool OUT_BF16, bool BIAS, bool GELU_ACT, bool RES>
__global__ void __launch_bounds__(256) gemm_bf16_kernel(
    const __bf16* __restrict__ A, const __bf16* __restrict__ B,
    __bf16* __restrict__ outb, float* __restrict__ outf,
    const float* __restrict__ bias, const float* __restrict__ res,
    int M, int N, int K) {
  __shared__ __bf16 As[2][BM * LDA];
  __shared__ __bf16 Bt[2][BN * LDBT];   // stored transposed: [n][k]
  int tid = threadIdx.x;
  int lane = tid & 31, wave = tid >> 5;
  int m0 = blockIdx.y * BM;
  int n0 = blockIdx.x * BN;
  int wr = (wave & 3) * 32;
  int wc = (wave >> 2) * 32;
  v8f acc00 = {}, acc01 = {}, acc10 = {}, acc11 = {};
  int bk = tid >> 3, bn = (tid & 7) * 8;     // B: 32 rows x 64

  // stage A tile for step `s` into buffer `buf`
  auto stage_A = [&](int buf, int k0) {
#if HAVE_TDM
    if (wave == 0)
      tdm_load_2d_pad((unsigned)(size_t)&As[buf][0],
                      A + (size_t)m0 * K + k0, BK, BM, K);
#else
#pragma unroll
    for (int it = 0; it < 2; ++it) {
      int chunk = tid + it * 256;
      int row = chunk >> 2, ck = (chunk & 3) * 8;
      async_copy16(A + (size_t)(m0 + row) * K + k0 + ck,
                   &As[buf][row * LDA + ck]);
    }
#endif
  };
  auto finish_stage = [&]() {
#if HAVE_TDM
    if (wave == 0) wait_tensor0();
#else
    wait_async0();
#endif
  };

  int nsteps = K / BK;
  // prologue: buffer 0
  stage_A(0, 0);
  {
    union { uint4 v; __bf16 h[8]; } bv;
    bv.v = *(const uint4*)(B + (size_t)bk * N + n0 + bn);
#pragma unroll
    for (int i = 0; i < 8; ++i) Bt[0][(bn + i) * LDBT + bk] = bv.h[i];
  }
  finish_stage();
  __syncthreads();

  for (int s = 0; s < nsteps; ++s) {
    int cur = s & 1, nxt = cur ^ 1;
    bool more = (s + 1) < nsteps;
    union { uint4 v; __bf16 h[8]; } bv;
    if (more) {
      stage_A(nxt, (s + 1) * BK);                     // TDM overlaps compute
      bv.v = *(const uint4*)(B + (size_t)((s + 1) * BK + bk) * N + n0 + bn);
      if (s + 2 < nsteps)
        __builtin_prefetch(B + (size_t)((s + 2) * BK + bk) * N + n0 + bn,
                           0, 0);
    }
    v16bf a0 = load_a_frag(&As[cur][wr * LDA], LDA);
    v16bf a1 = load_a_frag(&As[cur][(wr + 16) * LDA], LDA);
    v16bf b0 = load_b_frag(&Bt[cur][wc * LDBT], LDBT);
    v16bf b1 = load_b_frag(&Bt[cur][(wc + 16) * LDBT], LDBT);
    acc00 = wmma_bf16(a0, b0, acc00);
    acc01 = wmma_bf16(a0, b1, acc01);
    acc10 = wmma_bf16(a1, b0, acc10);
    acc11 = wmma_bf16(a1, b1, acc11);
    if (more) {
#pragma unroll
      for (int i = 0; i < 8; ++i) Bt[nxt][(bn + i) * LDBT + bk] = bv.h[i];
    }
    finish_stage();
    __syncthreads();
  }

  int nl = lane & 15, hf = lane >> 4;
#pragma unroll
  for (int mi = 0; mi < 2; ++mi) {
    v8f* am = mi ? &acc10 : &acc00;
    v8f* ap = mi ? &acc11 : &acc01;
#pragma unroll
    for (int r = 0; r < 8; ++r) {
      int m = m0 + wr + mi * 16 + r + hf * 8;
      int c0 = n0 + wc + nl;
      int c1 = c0 + 16;
      float v0 = (*am)[r], v1 = (*ap)[r];
      if (BIAS) { v0 += bias[c0]; v1 += bias[c1]; }
      if (GELU_ACT) { v0 = gelu_f(v0); v1 = gelu_f(v1); }
      if (RES) {
        v0 += res[(size_t)m * N + c0];
        v1 += res[(size_t)m * N + c1];
      }
      if (OUT_BF16) {
        outb[(size_t)m * N + c0] = (__bf16)v0;
        outb[(size_t)m * N + c1] = (__bf16)v1;
      } else {
        outf[(size_t)m * N + c0] = v0;
        outf[(size_t)m * N + c1] = v1;
      }
    }
  }
}

// ---------------------------------------------------------------------------
// Flash attention per (b,h): 128 Q rows / block, 8 waves x 16 rows, online
// softmax over 32-key chunks, double-buffered K/V staging (K via
// async-to-LDS, V via transpose-scatter).  Diagonal logits forced to 0.
// ---------------------------------------------------------------------------
#define LQ 72   // 64 + 8 pad
#define LK 72
#define LV 40   // 32 + 8 pad
#define LP 40

__global__ void __launch_bounds__(256) attn_kernel(
    const __bf16* __restrict__ qkv, __bf16* __restrict__ att) {
  int h = blockIdx.y, b = blockIdx.z;
  int q0 = blockIdx.x * 128;
  __shared__ __bf16 Qs[128 * LQ];
  __shared__ __bf16 Ks[2][32 * LK];     // [key][s] == B^T layout for QK^T
  __shared__ __bf16 Vt[2][64 * LV];     // [s][key] == B^T layout for PV
  __shared__ __bf16 Ps[8 * 16 * LP];    // per-wave P scratch (16x32)
  int tid = threadIdx.x, lane = tid & 31, wave = tid >> 5;
  size_t baseQ = (size_t)b * SEQ * C3 + (size_t)h * HEAD_SZ;
  int krow = tid >> 3, kc8 = (tid & 7) * 8;

  // prologue: Q tile (async) + K/V chunk 0
#pragma unroll
  for (int it = 0; it < 4; ++it) {
    int chunk = tid + it * 256;
    int row = chunk >> 3, c8 = (chunk & 7) * 8;
    async_copy16(qkv + baseQ + (size_t)(q0 + row) * C3 + c8,
                 Qs + row * LQ + c8);
  }
  {
    async_copy16(qkv + baseQ + D_MODEL + (size_t)krow * C3 + kc8,
                 &Ks[0][krow * LK + kc8]);
    union { uint4 v; __bf16 hx[8]; } vv;
    vv.v = *(const uint4*)(qkv + baseQ + 2 * D_MODEL + (size_t)krow * C3 + kc8);
#pragma unroll
    for (int i = 0; i < 8; ++i) Vt[0][(kc8 + i) * LV + krow] = vv.hx[i];
  }
  wait_async0();
  __syncthreads();

  int wr = wave * 16;
  v8f o0 = {}, o1 = {}, o2 = {}, o3 = {};
  float mrow[8], lrow[8];
#pragma unroll
  for (int r = 0; r < 8; ++r) { mrow[r] = -1e30f; lrow[r] = 0.f; }
  int nl = lane & 15, hf = lane >> 4;
  __bf16* Pw = Ps + wave * 16 * LP;

  const int NCHUNK = SEQ / 32;
  for (int s = 0; s < NCHUNK; ++s) {
    int u0 = s * 32;
    int cur = s & 1, nxt = cur ^ 1;
    bool more = (s + 1) < NCHUNK;
    union { uint4 v; __bf16 hx[8]; } vv;
    if (more) {  // stage next K (async) + next V (to regs) while computing
      async_copy16(qkv + baseQ + D_MODEL + (size_t)(u0 + 32 + krow) * C3 + kc8,
                   &Ks[nxt][krow * LK + kc8]);
      vv.v = *(const uint4*)(qkv + baseQ + 2 * D_MODEL +
                             (size_t)(u0 + 32 + krow) * C3 + kc8);
    }

    // S = Q * K^T : two 16x16 tiles per wave, K-dim 64 = 2 WMMA steps
    v8f s0 = {}, s1 = {};
#pragma unroll
    for (int kk = 0; kk < 64; kk += 32) {
      v16bf a = load_a_frag(Qs + wr * LQ + kk, LQ);
      v16bf b0 = load_b_frag(&Ks[cur][kk], LK);
      v16bf b1 = load_b_frag(&Ks[cur][16 * LK + kk], LK);
      s0 = wmma_bf16(a, b0, s0);
      s1 = wmma_bf16(a, b1, s1);
    }

    // scale (hs^-0.5 = 0.125), diagonal->0 mask, online softmax update
#pragma unroll
    for (int r = 0; r < 8; ++r) {
      int m = q0 + wr + r + hf * 8;
      float e0 = s0[r] * 0.125f, e1 = s1[r] * 0.125f;
      if (m == u0 + nl) e0 = 0.f;
      if (m == u0 + 16 + nl) e1 = 0.f;
      float mx = rmax16(fmaxf(e0, e1));
      float mn = fmaxf(mrow[r], mx);
      float corr = __expf(mrow[r] - mn);
      float p0 = __expf(e0 - mn), p1 = __expf(e1 - mn);
      lrow[r] = lrow[r] * corr + rsum16(p0 + p1);
      mrow[r] = mn;
      o0[r] *= corr; o1[r] *= corr; o2[r] *= corr; o3[r] *= corr;
      int ml = r + hf * 8;
      Pw[ml * LP + nl] = (__bf16)p0;
      Pw[ml * LP + 16 + nl] = (__bf16)p1;
    }

    // O += P(16x32) * V(32x64): 4 WMMAs
    v16bf pa = load_a_frag(Pw, LP);
    o0 = wmma_bf16(pa, load_b_frag(&Vt[cur][0 * 16 * LV], LV), o0);
    o1 = wmma_bf16(pa, load_b_frag(&Vt[cur][1 * 16 * LV], LV), o1);
    o2 = wmma_bf16(pa, load_b_frag(&Vt[cur][2 * 16 * LV], LV), o2);
    o3 = wmma_bf16(pa, load_b_frag(&Vt[cur][3 * 16 * LV], LV), o3);

    if (more) {
#pragma unroll
      for (int i = 0; i < 8; ++i) Vt[nxt][(kc8 + i) * LV + krow] = vv.hx[i];
    }
    wait_async0();
    __syncthreads();
  }

  // normalize and emit concat-head layout [B*T, C]
#pragma unroll
  for (int r = 0; r < 8; ++r) {
    int t = q0 + wr + r + hf * 8;
    float inv = lrow[r] > 0.f ? 1.0f / lrow[r] : 0.f;
    size_t base = ((size_t)b * SEQ + t) * D_MODEL + h * HEAD_SZ;
    att[base + 0 * 16 + nl] = (__bf16)(o0[r] * inv);
    att[base + 1 * 16 + nl] = (__bf16)(o1[r] * inv);
    att[base + 2 * 16 + nl] = (__bf16)(o2[r] * inv);
    att[base + 3 * 16 + nl] = (__bf16)(o3[r] * inv);
  }
}

// ---------------------------------------------------------------------------
// Host-side orchestration
// ---------------------------------------------------------------------------
extern "C" void kernel_launch(void* const* d_in, const int* in_sizes, int n_in,
                              void* d_out, int out_size, void* d_ws,
                              size_t ws_size, hipStream_t stream) {
  const float* x   = (const float*)d_in[0];
  const float* Wq  = (const float*)d_in[1];
  const float* Wk  = (const float*)d_in[2];
  const float* Wv  = (const float*)d_in[3];
  const float* Wo  = (const float*)d_in[4];
  const float* bo  = (const float*)d_in[5];
  const float* W1  = (const float*)d_in[6];
  const float* b1  = (const float*)d_in[7];
  const float* W2  = (const float*)d_in[8];
  const float* b2  = (const float*)d_in[9];
  const float* g1  = (const float*)d_in[10];
  const float* be1 = (const float*)d_in[11];
  const float* g2  = (const float*)d_in[12];
  const float* be2 = (const float*)d_in[13];
  float* out = (float*)d_out;

  char* ws = (char*)d_ws;
  size_t off = 0;
  auto take = [&](size_t bytes) -> char* {
    char* p = ws + off;
    off = (off + bytes + 255) & ~(size_t)255;
    return p;
  };
  __bf16* Wqkv_b = (__bf16*)take((size_t)D_MODEL * C3 * 2);
  __bf16* Wo_b   = (__bf16*)take((size_t)D_MODEL * D_MODEL * 2);
  __bf16* W1_b   = (__bf16*)take((size_t)D_MODEL * FF * 2);
  __bf16* W2_b   = (__bf16*)take((size_t)FF * D_MODEL * 2);
  __bf16* h_ln   = (__bf16*)take((size_t)MROWS * D_MODEL * 2);
  __bf16* qkv    = (__bf16*)take((size_t)MROWS * C3 * 2);
  __bf16* attb   = (__bf16*)take((size_t)MROWS * D_MODEL * 2);
  float*  x1     = (float*)take((size_t)MROWS * D_MODEL * 4);
  __bf16* h2     = (__bf16*)take((size_t)MROWS * D_MODEL * 2);
  __bf16* ff1    = (__bf16*)take((size_t)MROWS * FF * 2);

  // 1) repack weights to bf16
  pack_qkv_kernel<<<(D_MODEL * C3) / 256, 256, 0, stream>>>(Wq, Wk, Wv, Wqkv_b);
  cvt_bf16_kernel<<<(D_MODEL * D_MODEL) / 256, 256, 0, stream>>>(
      Wo, Wo_b, D_MODEL * D_MODEL);
  cvt_bf16_kernel<<<(D_MODEL * FF) / 256, 256, 0, stream>>>(W1, W1_b,
                                                            D_MODEL * FF);
  cvt_bf16_kernel<<<(FF * D_MODEL) / 256, 256, 0, stream>>>(W2, W2_b,
                                                            FF * D_MODEL);

  // 2) LN1 -> bf16 activations
  ln_kernel<<<MROWS, 256, 0, stream>>>(x, g1, be1, h_ln);

  // 3) packed QKV GEMM  [8192 x 1024] * [1024 x 3072] -> bf16
  gemm_bf16_kernel<true, false, false, false>
      <<<dim3(C3 / BN, MROWS / BM), 256, 0, stream>>>(
          h_ln, Wqkv_b, qkv, nullptr, nullptr, nullptr, MROWS, C3, D_MODEL);

  // 4) attention (flash, per b,h, 128-row Q tiles)
  attn_kernel<<<dim3(SEQ / 128, N_HEAD, BATCH), 256, 0, stream>>>(qkv, attb);

  // 5) output projection + bias + residual -> x1 (fp32)
  gemm_bf16_kernel<false, true, false, true>
      <<<dim3(D_MODEL / BN, MROWS / BM), 256, 0, stream>>>(
          attb, Wo_b, nullptr, x1, bo, x, MROWS, D_MODEL, D_MODEL);

  // 6) LN2 -> bf16
  ln_kernel<<<MROWS, 256, 0, stream>>>(x1, g2, be2, h2);

  // 7) FFN1 + bias + exact GELU -> bf16 [8192 x 4096]
  gemm_bf16_kernel<true, true, true, false>
      <<<dim3(FF / BN, MROWS / BM), 256, 0, stream>>>(
          h2, W1_b, ff1, nullptr, b1, nullptr, MROWS, FF, D_MODEL);

  // 8) FFN2 + bias + residual -> d_out (fp32)
  gemm_bf16_kernel<false, true, false, true>
      <<<dim3(D_MODEL / BN, MROWS / BM), 256, 0, stream>>>(
          ff1, W2_b, nullptr, out, b2, x1, MROWS, D_MODEL, FF);
}
